// Qwen3MoeDecoderLayer_2791728742952
// MI455X (gfx1250) — matrix operand order repaired and checked by reading, not compiled, not run on hardware
//
#include <hip/hip_runtime.h>
#include <stdint.h>

typedef __bf16 bf16;
typedef __attribute__((ext_vector_type(4)))  __bf16 v4bf;
typedef __attribute__((ext_vector_type(8)))  __bf16 v8bf;
typedef __attribute__((ext_vector_type(16))) __bf16 v16bf;
typedef __attribute__((ext_vector_type(4)))  float  v4f;
typedef __attribute__((ext_vector_type(8)))  float  v8f;

#define T_   1024
#define H_   2048
#define NH_  16
#define NKV_ 4
#define HD_  128
#define E_   8
#define I_   768
#define QKV_ 3072

// ---------------------------------------------------------------- CDNA5 async copy
// GLOBAL_LOAD_ASYNC_TO_LDS_B128: per-lane 16B global->LDS, tracked by ASYNCcnt.
__device__ __forceinline__ void async_load_b128(unsigned lds_addr, const void* gptr) {
  asm volatile("global_load_async_to_lds_b128 %0, %1, off"
               :
               : "v"(lds_addr), "v"((unsigned long long)(size_t)gptr)
               : "memory");
}
__device__ __forceinline__ void wait_async0() {
  asm volatile("s_wait_asynccnt 0x0" ::: "memory");
}

// ---------------------------------------------------------------- fragments
// 16-bit A-frag (16x32, MxK): lane L: m=L&15, khalf=(L>>4)*16;
//   elems = A[m][khalf..khalf+15]
// 16-bit B-frag (32x16, KxN): lane L: n=L&15, khalf=(L>>4)*16;
//   elems = B[khalf..khalf+15][n]  (B kept transposed [n][k] in LDS)
__device__ __forceinline__ v16bf frag16(const bf16* lo, const bf16* hi) {
  v8bf a = *(const v8bf*)lo;
  v8bf b = *(const v8bf*)hi;
  v16bf r;
#pragma unroll
  for (int i = 0; i < 8; ++i) { r[i] = a[i]; r[i + 8] = b[i]; }
  return r;
}

__device__ __forceinline__ v8f wmma_bf16(v16bf a, v16bf b, v8f c) {
  return __builtin_amdgcn_wmma_f32_16x16x32_bf16(false, a, false, b,
                                                 (short)0, c, false, false);
}

// ---------------------------------------------------------------- utilities
__global__ __launch_bounds__(256) void cvt_bf16_k(const float* __restrict__ in,
                                                  bf16* __restrict__ out, long n4) {
  long i = (long)blockIdx.x * blockDim.x + threadIdx.x;
  long st = (long)gridDim.x * blockDim.x;
  for (; i < n4; i += st) {
    v4f x = ((const v4f*)in)[i];
    v4bf y;
#pragma unroll
    for (int j = 0; j < 4; ++j) y[j] = (bf16)x[j];
    ((v4bf*)out)[i] = y;
  }
}

__global__ __launch_bounds__(256) void zero_f32_k(float* __restrict__ p, long n4) {
  long i = (long)blockIdx.x * blockDim.x + threadIdx.x;
  long st = (long)gridDim.x * blockDim.x;
  v4f z;
#pragma unroll
  for (int j = 0; j < 4; ++j) z[j] = 0.f;
  for (; i < n4; i += st) ((v4f*)p)[i] = z;
}

__global__ __launch_bounds__(64) void zero_i32_k(int* __restrict__ p, int n) {
  int i = threadIdx.x;
  if (i < n) p[i] = 0;
}

// ---------------------------------------------------------------- add+RMSNorm
__global__ __launch_bounds__(256)
void fused_add_rmsnorm(const float* __restrict__ a, const float* __restrict__ b,
                       const float* __restrict__ w, float* __restrict__ res,
                       bf16* __restrict__ hn) {
  const int t = blockIdx.x, tid = threadIdx.x;
  const long base = (long)t * H_ + tid * 8;
  __shared__ float red[256];
  v4f v0 = *(const v4f*)(a + base) + *(const v4f*)(b + base);
  v4f v1 = *(const v4f*)(a + base + 4) + *(const v4f*)(b + base + 4);
  *(v4f*)(res + base) = v0;
  *(v4f*)(res + base + 4) = v1;
  float loc = 0.f;
#pragma unroll
  for (int j = 0; j < 4; ++j) loc += v0[j] * v0[j] + v1[j] * v1[j];
  red[tid] = loc;
  __syncthreads();
  for (int s = 128; s > 0; s >>= 1) {
    if (tid < s) red[tid] += red[tid + s];
    __syncthreads();
  }
  const float rs = rsqrtf(red[0] / (float)H_ + 1e-6f);
  v4f w0 = *(const v4f*)(w + tid * 8);
  v4f w1 = *(const v4f*)(w + tid * 8 + 4);
  v4bf h0, h1;
#pragma unroll
  for (int j = 0; j < 4; ++j) {
    h0[j] = (bf16)(v0[j] * rs * w0[j]);
    h1[j] = (bf16)(v1[j] * rs * w1[j]);
  }
  *(v4bf*)(hn + base) = h0;
  *(v4bf*)(hn + base + 4) = h1;
}

// ---------------------------------------------------------------- generic WMMA GEMM
// C[M,N] (+)= A[M,K](bf16) x B[K,N](bf16)
// mode 0: f32 store, mode 1: bf16 store, mode 2: atomicAdd(out[ridx[m]], sw[m]*acc)
// a_gather: A row = ridx[m] (else compact row m). cnts[z] gives real M (device).
// A tile staged via ASYNC global->LDS DMA; B tile via register transpose.
// LDS double-buffered -> one barrier per K-stage. B-frag loads pipelined 1 ahead.
#define BM 128
#define BN 128
#define BK 64
__global__ __launch_bounds__(256)
void gemm_bf16_wmma(const bf16* __restrict__ A, int lda, long sA,
                    const bf16* __restrict__ B, int ldb, long sB,
                    void* __restrict__ Cv, int ldc, long sC,
                    int M, int N, int K,
                    const int* __restrict__ ridx, long sIdx,
                    const int* __restrict__ cnts,
                    const float* __restrict__ sw, long sW,
                    int mode, int a_gather) {
  const int z = blockIdx.z;
  const int Mreal = cnts ? cnts[z] : M;
  const int row0 = blockIdx.y * BM;
  if (row0 >= Mreal) return;
  const int col0 = blockIdx.x * BN;
  const bf16* Ab = A + (long)z * sA;
  const bf16* Bb = B + (long)z * sB;
  const int* ridxb = ridx ? ridx + (long)z * sIdx : nullptr;
  const float* swb = sw ? sw + (long)z * sW : nullptr;

  __shared__ bf16 As[2][BM][BK + 8];
  __shared__ bf16 Bs[2][BN][BK + 8];

  const int tid = threadIdx.x;
  const int wave = tid >> 5, lane = tid & 31;
  const int lhalf = lane >> 4, l16 = lane & 15;

  // staging geometry: 16B granularity, loop-invariant rows/pointers
  const int aseg = (tid & 7) * 8;    // k offset within A tile
  const int bn   = (tid & 15) * 8;   // n offset within B tile
  int am[4], bk[4];
  const bf16* pA[4];
  const bf16* pB[4];
  unsigned laA[2][4];                // LDS byte addresses for async A staging
#pragma unroll
  for (int it = 0; it < 4; ++it) {
    am[it] = (tid >> 3) + it * 32;   // A tile row
    bk[it] = (tid >> 4) + it * 16;   // B tile k row
    int gm = row0 + am[it];
    if (gm >= Mreal) gm = Mreal - 1;
    const int grow = (a_gather && ridxb) ? ridxb[gm] : gm;
    pA[it] = Ab + (long)grow * lda + aseg;
    pB[it] = Bb + (long)bk[it] * ldb + col0 + bn;
    laA[0][it] = (unsigned)(size_t)&As[0][am[it]][aseg];
    laA[1][it] = (unsigned)(size_t)&As[1][am[it]][aseg];
  }

  v8f acc[8];
#pragma unroll
  for (int i = 0; i < 8; ++i)
#pragma unroll
    for (int j = 0; j < 8; ++j) acc[i][j] = 0.f;

  // prologue: async A tile 0, register B tile 0
  v8bf rb[4];
#pragma unroll
  for (int it = 0; it < 4; ++it) async_load_b128(laA[0][it], pA[it]);
#pragma unroll
  for (int it = 0; it < 4; ++it) rb[it] = *(const v8bf*)pB[it];

  int buf = 0;
  for (int k0 = 0; k0 < K; k0 += BK) {
    // commit B registers (transposed) into current buffer
#pragma unroll
    for (int it = 0; it < 4; ++it)
#pragma unroll
      for (int j = 0; j < 8; ++j) Bs[buf][bn + j][bk[it]] = rb[it][j];
    wait_async0();        // our async A writes for this buffer have landed
    __syncthreads();      // everyone's A DMA + B stores visible
    if (k0 + BK < K) {    // kick off next tile into the other buffer
#pragma unroll
      for (int it = 0; it < 4; ++it) {
        pA[it] += BK;
        async_load_b128(laA[buf ^ 1][it], pA[it]);
        pB[it] += (long)BK * ldb;
        rb[it] = *(const v8bf*)pB[it];
      }
    }
    const int m = wave * 16 + l16;
    // hoist both A fragments; pipeline B fragments one WMMA ahead
    v16bf afr[2];
    afr[0] = frag16(&As[buf][m][lhalf * 16], &As[buf][m][lhalf * 16 + 8]);
    afr[1] = frag16(&As[buf][m][32 + lhalf * 16], &As[buf][m][32 + lhalf * 16 + 8]);
    v16bf curb = frag16(&Bs[buf][l16][lhalf * 16], &Bs[buf][l16][lhalf * 16 + 8]);
#pragma unroll
    for (int kc = 0; kc < 2; ++kc) {
#pragma unroll
      for (int nt = 0; nt < 8; ++nt) {
        v16bf nb = curb;
        if (!(kc == 1 && nt == 7)) {
          const int nkc = (nt == 7) ? 1 : kc;
          const int nnt = (nt == 7) ? 0 : nt + 1;
          const int n = nnt * 16 + l16;
          nb = frag16(&Bs[buf][n][nkc * 32 + lhalf * 16],
                      &Bs[buf][n][nkc * 32 + lhalf * 16 + 8]);
        }
        acc[nt] = wmma_bf16(afr[kc], curb, acc[nt]);
        curb = nb;
      }
    }
    buf ^= 1;
  }

#pragma unroll
  for (int nt = 0; nt < 8; ++nt) {
#pragma unroll
    for (int r = 0; r < 8; ++r) {
      const int m = row0 + wave * 16 + r + lhalf * 8;
      if (m >= Mreal) continue;
      const int n = col0 + nt * 16 + l16;
      const float v = acc[nt][r];
      if (mode == 0) {
        ((float*)Cv + (long)z * sC)[(long)m * ldc + n] = v;
      } else if (mode == 1) {
        ((bf16*)Cv + (long)z * sC)[(long)m * ldc + n] = (bf16)v;
      } else {
        const int g = ridxb ? ridxb[m] : m;
        const float wt = swb ? swb[m] : 1.f;
        atomicAdd((float*)Cv + (long)g * ldc + n, wt * v);
      }
    }
  }
}

// ---------------------------------------------------------------- QK norm + RoPE
__global__ __launch_bounds__(128)
void qk_rope(const float* __restrict__ qkv, const int* __restrict__ pos,
             const float* __restrict__ qw, const float* __restrict__ kw,
             bf16* __restrict__ Qo, bf16* __restrict__ Ko, bf16* __restrict__ Vo) {
  const int t = blockIdx.x, y = blockIdx.y, d = threadIdx.x;
  if (y >= 20) {  // plain V convert
    int kvh = y - 20;
    Vo[(long)t * (NKV_ * HD_) + kvh * HD_ + d] =
        (bf16)qkv[(long)t * QKV_ + NH_ * HD_ + NKV_ * HD_ + kvh * HD_ + d];
    return;
  }
  __shared__ float red[128];
  __shared__ float nv[128];
  const float* src;
  const float* w;
  if (y < 16) { src = qkv + (long)t * QKV_ + y * HD_; w = qw; }
  else        { src = qkv + (long)t * QKV_ + NH_ * HD_ + (y - 16) * HD_; w = kw; }
  float x = src[d];
  red[d] = x * x;
  __syncthreads();
  for (int s = 64; s > 0; s >>= 1) {
    if (d < s) red[d] += red[d + s];
    __syncthreads();
  }
  float rs = rsqrtf(red[0] / (float)HD_ + 1e-6f);
  float n = x * rs * w[d];
  nv[d] = n;
  __syncthreads();
  float p = (float)pos[t];
  int i = d & 63;
  float freq = __expf(-(float)(2 * i) / (float)HD_ * 13.815510558f);  // ln(1e6)
  float ang = p * freq, sn, cs;
  __sincosf(ang, &sn, &cs);
  float out = (d < 64) ? (n * cs - nv[d + 64] * sn) : (n * cs + nv[d - 64] * sn);
  if (y < 16) Qo[(long)t * (NH_ * HD_) + y * HD_ + d] = (bf16)out;
  else        Ko[(long)t * (NKV_ * HD_) + (y - 16) * HD_ + d] = (bf16)out;
}

// ---------------------------------------------------------------- flash attention
// one wave per (head, 16-query block); 32 kv per iteration via two 16x16 S tiles
__global__ __launch_bounds__(32)
void attn_kernel(const bf16* __restrict__ Q, const bf16* __restrict__ Kb,
                 const bf16* __restrict__ Vb, bf16* __restrict__ O) {
  const int h = blockIdx.x;
  const int qb = blockIdx.y;
  const int kvh = h >> 2;  // NH/NKV = 4
  const int lane = threadIdx.x;
  const int lhalf = lane >> 4, l16 = lane & 15;
  __shared__ bf16 Qs[16][128];
  __shared__ bf16 Ks[32][128];
  __shared__ bf16 Vt[128][32];  // transposed [feat][kv]
  __shared__ bf16 Ps[16][32];

  const int q0 = qb * 16;
  for (int u = lane; u < 16 * 16; u += 32) {  // 16B units
    int r = u >> 4, seg = (u & 15) * 8;
    *(v8bf*)&Qs[r][seg] =
        *(const v8bf*)&Q[(long)(q0 + r) * (NH_ * HD_) + h * HD_ + seg];
  }

  float rm[8], ls[8];
  v8f accO[8];
#pragma unroll
  for (int r = 0; r < 8; ++r) { rm[r] = -3.0e38f; ls[r] = 0.f; }
#pragma unroll
  for (int nt = 0; nt < 8; ++nt)
#pragma unroll
    for (int r = 0; r < 8; ++r) accO[nt][r] = 0.f;

  const float scale = 0.08838834764831845f;  // 1/sqrt(128)
  const int jmax = (q0 + 15) >> 5;
  for (int j = 0; j <= jmax; ++j) {
    const int kv0 = j * 32;
    for (int u = lane; u < 32 * 16; u += 32) {
      int r = u >> 4, seg = (u & 15) * 8;
      long base = (long)(kv0 + r) * (NKV_ * HD_) + kvh * HD_ + seg;
      async_load_b128((unsigned)(size_t)&Ks[r][seg], &Kb[base]);  // K: async DMA
      v8bf vv = *(const v8bf*)&Vb[base];
#pragma unroll
      for (int jj = 0; jj < 8; ++jj) Vt[seg + jj][r] = vv[jj];
    }
    wait_async0();
    __syncthreads();

    v8f S0, S1;
#pragma unroll
    for (int r = 0; r < 8; ++r) { S0[r] = 0.f; S1[r] = 0.f; }
#pragma unroll
    for (int c = 0; c < 4; ++c) {
      v16bf aq = frag16(&Qs[l16][c * 32 + lhalf * 16],
                        &Qs[l16][c * 32 + lhalf * 16 + 8]);
      v16bf b0 = frag16(&Ks[l16][c * 32 + lhalf * 16],
                        &Ks[l16][c * 32 + lhalf * 16 + 8]);
      v16bf b1 = frag16(&Ks[16 + l16][c * 32 + lhalf * 16],
                        &Ks[16 + l16][c * 32 + lhalf * 16 + 8]);
      S0 = wmma_bf16(aq, b0, S0);
      S1 = wmma_bf16(aq, b1, S1);
    }

    float corr[8];
#pragma unroll
    for (int r = 0; r < 8; ++r) {
      const int rowg = q0 + r + lhalf * 8;
      const int c0 = kv0 + l16, c1 = c0 + 16;
      float s0 = S0[r] * scale; if (c0 > rowg) s0 = -3.0e38f;
      float s1 = S1[r] * scale; if (c1 > rowg) s1 = -3.0e38f;
      float bm = fmaxf(s0, s1);
      for (int msk = 1; msk < 16; msk <<= 1) bm = fmaxf(bm, __shfl_xor(bm, msk, 32));
      float mnew = fmaxf(rm[r], bm);
      float p0 = __expf(s0 - mnew), p1 = __expf(s1 - mnew);
      float ps = p0 + p1;
      for (int msk = 1; msk < 16; msk <<= 1) ps += __shfl_xor(ps, msk, 32);
      float cr = __expf(rm[r] - mnew);
      ls[r] = ls[r] * cr + ps;
      rm[r] = mnew;
      corr[r] = cr;
      Ps[r + lhalf * 8][l16] = (bf16)p0;
      Ps[r + lhalf * 8][16 + l16] = (bf16)p1;
    }
#pragma unroll
    for (int nt = 0; nt < 8; ++nt)
#pragma unroll
      for (int r = 0; r < 8; ++r) accO[nt][r] *= corr[r];
    __syncthreads();
    v16bf ap = frag16(&Ps[l16][lhalf * 16], &Ps[l16][lhalf * 16 + 8]);
#pragma unroll
    for (int nt = 0; nt < 8; ++nt) {
      const int f = nt * 16 + l16;
      v16bf bv = frag16(&Vt[f][lhalf * 16], &Vt[f][lhalf * 16 + 8]);
      accO[nt] = wmma_bf16(ap, bv, accO[nt]);
    }
    __syncthreads();
  }

#pragma unroll
  for (int nt = 0; nt < 8; ++nt)
#pragma unroll
    for (int r = 0; r < 8; ++r)
      O[(long)(q0 + r + lhalf * 8) * (NH_ * HD_) + h * HD_ + nt * 16 + l16] =
          (bf16)(accO[nt][r] / ls[r]);
}

// ---------------------------------------------------------------- MoE router
__global__ __launch_bounds__(256)
void router_kernel(const bf16* __restrict__ hn, const float* __restrict__ gw,
                   int* __restrict__ cnt, int* __restrict__ idx,
                   float* __restrict__ wgt) {
  const int t = blockIdx.x, tid = threadIdx.x;
  const int e = tid & 7, ch = tid >> 3;  // 32 chunks of 64
  float p = 0.f;
  for (int k = ch * 64; k < ch * 64 + 64; ++k)
    p += (float)hn[(long)t * H_ + k] * gw[k * E_ + e];
  __shared__ float red[256];
  red[tid] = p;
  __syncthreads();
  if (tid < 8) {
    float s = 0.f;
    for (int c = 0; c < 32; ++c) s += red[c * 8 + tid];
    red[tid] = s;
  }
  __syncthreads();
  if (tid == 0) {
    float lg[E_], mx = -3.0e38f;
    for (int i = 0; i < E_; ++i) { lg[i] = red[i]; mx = fmaxf(mx, lg[i]); }
    float sm = 0.f;
    for (int i = 0; i < E_; ++i) { lg[i] = __expf(lg[i] - mx); sm += lg[i]; }
    for (int i = 0; i < E_; ++i) lg[i] /= sm;
    int i1 = 0;
    for (int i = 1; i < E_; ++i) if (lg[i] > lg[i1]) i1 = i;
    int i2 = (i1 == 0) ? 1 : 0;
    for (int i = 0; i < E_; ++i) if (i != i1 && lg[i] > lg[i2]) i2 = i;
    float tot = lg[i1] + lg[i2];
    int p1 = atomicAdd(&cnt[i1], 1);
    idx[i1 * T_ + p1] = t; wgt[i1 * T_ + p1] = lg[i1] / tot;
    int p2 = atomicAdd(&cnt[i2], 1);
    idx[i2 * T_ + p2] = t; wgt[i2 * T_ + p2] = lg[i2] / tot;
  }
}

// ---------------------------------------------------------------- SiLU * up
__global__ __launch_bounds__(256)
void silu_mul(const bf16* __restrict__ gu, bf16* __restrict__ act,
              const int* __restrict__ cnt) {
  const int e = blockIdx.z, m = blockIdx.y;
  if (m >= cnt[e]) return;
  const int i = blockIdx.x * 256 + threadIdx.x;
  const long base = ((long)e * T_ + m) * (2 * I_);
  float g = (float)gu[base + i];
  float u = (float)gu[base + I_ + i];
  float s = g / (1.f + __expf(-g));
  act[((long)e * T_ + m) * I_ + i] = (bf16)(s * u);
}

// ---------------------------------------------------------------- launch
extern "C" void kernel_launch(void* const* d_in, const int* in_sizes, int n_in,
                              void* d_out, int out_size, void* d_ws, size_t ws_size,
                              hipStream_t stream) {
  const int*   positions = (const int*)d_in[0];
  const float* hidden    = (const float*)d_in[1];
  const float* residual  = (const float*)d_in[2];
  const float* w_qkv     = (const float*)d_in[3];
  const float* w_o       = (const float*)d_in[4];
  const float* q_norm_w  = (const float*)d_in[5];
  const float* k_norm_w  = (const float*)d_in[6];
  const float* ln1_w     = (const float*)d_in[7];
  const float* ln2_w     = (const float*)d_in[8];
  const float* gate_w    = (const float*)d_in[9];
  const float* w_gate_up = (const float*)d_in[10];
  const float* w_down    = (const float*)d_in[11];
  float* out = (float*)d_out;  // [T*H] moe-out, then [T*H] residual

  char* cur = (char*)d_ws;
  auto alloc = [&](size_t bytes) -> void* {
    void* p = (void*)cur;
    cur += (bytes + 255) & ~(size_t)255;
    return p;
  };
  bf16* wqkv_b = (bf16*)alloc((size_t)H_ * QKV_ * 2);
  bf16* wo_b   = (bf16*)alloc((size_t)H_ * H_ * 2);
  bf16* wgu_b  = (bf16*)alloc((size_t)E_ * H_ * 2 * I_ * 2);
  bf16* wdn_b  = (bf16*)alloc((size_t)E_ * I_ * H_ * 2);
  float* res1  = (float*)alloc((size_t)T_ * H_ * 4);
  bf16* hn1    = (bf16*)alloc((size_t)T_ * H_ * 2);
  float* qkvf  = (float*)alloc((size_t)T_ * QKV_ * 4);
  bf16* qr     = (bf16*)alloc((size_t)T_ * NH_ * HD_ * 2);
  bf16* kr     = (bf16*)alloc((size_t)T_ * NKV_ * HD_ * 2);
  bf16* vr     = (bf16*)alloc((size_t)T_ * NKV_ * HD_ * 2);
  bf16* obf    = (bf16*)alloc((size_t)T_ * H_ * 2);
  float* attnf = (float*)alloc((size_t)T_ * H_ * 4);
  bf16* hn2    = (bf16*)alloc((size_t)T_ * H_ * 2);
  int*  cnt    = (int*)alloc(E_ * 4);
  int*  idx    = (int*)alloc((size_t)E_ * T_ * 4);
  float* wgt   = (float*)alloc((size_t)E_ * T_ * 4);
  bf16* gu     = (bf16*)alloc((size_t)E_ * T_ * 2 * I_ * 2);
  bf16* act    = (bf16*)alloc((size_t)E_ * T_ * I_ * 2);
  (void)ws_size; (void)in_sizes; (void)n_in; (void)out_size;

  // init: zero moe-out accumulator + expert counters
  zero_f32_k<<<2048, 256, 0, stream>>>(out, (long)T_ * H_ / 4);
  zero_i32_k<<<1, 64, 0, stream>>>(cnt, E_);

  // weight conversion f32 -> bf16 (16B vector in / 8B out)
  cvt_bf16_k<<<2048, 256, 0, stream>>>(w_qkv, wqkv_b, (long)H_ * QKV_ / 4);
  cvt_bf16_k<<<2048, 256, 0, stream>>>(w_o, wo_b, (long)H_ * H_ / 4);
  cvt_bf16_k<<<4096, 256, 0, stream>>>(w_gate_up, wgu_b, (long)E_ * H_ * 2 * I_ / 4);
  cvt_bf16_k<<<4096, 256, 0, stream>>>(w_down, wdn_b, (long)E_ * I_ * H_ / 4);

  // 1) residual add + RMSNorm (ln1)
  fused_add_rmsnorm<<<T_, 256, 0, stream>>>(hidden, residual, ln1_w, res1, hn1);

  // 2) QKV projection
  gemm_bf16_wmma<<<dim3(QKV_ / BN, T_ / BM, 1), 256, 0, stream>>>(
      hn1, H_, 0, wqkv_b, QKV_, 0, qkvf, QKV_, 0, T_, QKV_, H_,
      nullptr, 0, nullptr, nullptr, 0, /*mode*/0, /*a_gather*/0);

  // 3) QK RMSNorm + RoPE (+ V convert)
  qk_rope<<<dim3(T_, 24), 128, 0, stream>>>(qkvf, positions, q_norm_w, k_norm_w,
                                            qr, kr, vr);

  // 4) causal GQA flash attention
  attn_kernel<<<dim3(NH_, T_ / 16), 32, 0, stream>>>(qr, kr, vr, obf);

  // 5) O projection
  gemm_bf16_wmma<<<dim3(H_ / BN, T_ / BM, 1), 256, 0, stream>>>(
      obf, H_, 0, wo_b, H_, 0, attnf, H_, 0, T_, H_, H_,
      nullptr, 0, nullptr, nullptr, 0, 0, 0);

  // 6) residual add + RMSNorm (ln2); residual goes straight to output slot 2
  fused_add_rmsnorm<<<T_, 256, 0, stream>>>(attnf, res1, ln2_w,
                                            out + (size_t)T_ * H_, hn2);

  // 7) router: softmax -> top-2 -> renorm -> per-expert gather lists
  router_kernel<<<T_, 256, 0, stream>>>(hn2, gate_w, cnt, idx, wgt);

  // 8) gate_up for routed tokens only (row-gathered A), bf16 out
  gemm_bf16_wmma<<<dim3(2 * I_ / BN, T_ / BM, E_), 256, 0, stream>>>(
      hn2, H_, 0, wgu_b, 2 * I_, (long)H_ * 2 * I_, gu, 2 * I_, (long)T_ * 2 * I_,
      T_, 2 * I_, H_, idx, T_, cnt, nullptr, 0, /*mode*/1, /*a_gather*/1);

  // 9) SiLU(g) * u
  silu_mul<<<dim3(I_ / 256, T_, E_), 256, 0, stream>>>(gu, act, cnt);

  // 10) down projection + weighted scatter-add into out
  gemm_bf16_wmma<<<dim3(H_ / BN, T_ / BM, E_), 256, 0, stream>>>(
      act, I_, (long)T_ * I_, wdn_b, H_, (long)I_ * H_, out, H_, 0,
      T_, H_, I_, idx, T_, cnt, wgt, T_, /*mode*/2, /*a_gather*/0);
}
